// QSMPBlock_59906203844676
// MI455X (gfx1250) — matrix operand order repaired and checked by reading, not compiled, not run on hardware
//
#include <hip/hip_runtime.h>
#include <hip/hip_bf16.h>
#include <math.h>

// MI455X / gfx1250 implementation.
// bf16 + v_wmma_f32_16x16x32_bf16 for all GEMMs and flash attention
// (scores never materialized). Tile staging uses the CDNA5 Tensor Data Mover
// (tensor_load_to_lds + s_wait_tensorcnt) with D# pad fields producing the
// padded LDS row strides directly.

typedef __bf16 bf16_t;
typedef __attribute__((ext_vector_type(8)))  __bf16 v8bf;
typedef __attribute__((ext_vector_type(16))) __bf16 v16bf;
typedef __attribute__((ext_vector_type(8)))  float  v8f;
typedef __attribute__((ext_vector_type(4)))  unsigned u32x4;
typedef __attribute__((ext_vector_type(8)))  int      i32x8;
typedef __attribute__((ext_vector_type(4)))  int      i32x4;

#define BB 2
#define NN 2048
#define DD 512
#define HH 8
#define HD 64
#define BN (BB*NN)

static __device__ __forceinline__ float wred_sum16(float v){
  v += __shfl_xor(v, 1, 32); v += __shfl_xor(v, 2, 32);
  v += __shfl_xor(v, 4, 32); v += __shfl_xor(v, 8, 32);
  return v;
}
static __device__ __forceinline__ float wred_max16(float v){
  v = fmaxf(v, __shfl_xor(v, 1, 32)); v = fmaxf(v, __shfl_xor(v, 2, 32));
  v = fmaxf(v, __shfl_xor(v, 4, 32)); v = fmaxf(v, __shfl_xor(v, 8, 32));
  return v;
}
static __device__ __forceinline__ float wred_sum32(float v){
  v += __shfl_xor(v, 16, 32); return wred_sum16(v);
}

// ---------------------------------------------------------------------------
// Tensor Data Mover: 2D bf16 tile load global->LDS with per-row LDS padding.
//   tile_w elements per row, tile_h rows, global row stride `stride` elements.
//   pad_icode: DWORDs-per-row code (2<<code DWORDs), pad_acode: pad DWORDs-1.
// D# layout per CDNA5 ISA §8.3-8.6 (group2/3 zero => 2-D tensor).
// amdgpu-toolchain builtin arity: (u32x4, i32x8, i32x4, i32x4, i32x8, i32).
// ---------------------------------------------------------------------------
static __device__ __forceinline__ void tdm_load_2d(
    const void* gaddr, unsigned lds_off, int tile_w, int tile_h,
    long stride_elems, int pad_icode, int pad_acode)
{
  unsigned long long ga = (unsigned long long)(size_t)gaddr;
  u32x4 g0;
  g0[0] = 1u;                                           // count=1 (valid), user
  g0[1] = lds_off;                                      // lds_addr (bytes)
  g0[2] = (unsigned)(ga & 0xffffffffu);                 // global_addr[31:0]
  g0[3] = (unsigned)((ga >> 32) & 0x1ffffffu)           // global_addr[56:32]
        | (2u << 30);                                   // type = 2 ("image")
  unsigned td0 = (unsigned)stride_elems;                // tensor_dim0 (>= tile_w)
  unsigned td1 = 0x100000u;                             // tensor_dim1 (large)
  i32x8 g1;
  g1[0] = (int)((1u << 16)                              // data_size = 2 bytes
        | (1u << 20)                                    // pad_enable
        | ((unsigned)pad_icode << 22)
        | ((unsigned)pad_acode << 25));
  g1[1] = (int)((td0 & 0xffffu) << 16);                 // tensor_dim0[15:0]
  g1[2] = (int)((td0 >> 16) | ((td1 & 0xffffu) << 16)); // td0[31:16], td1[15:0]
  g1[3] = (int)((td1 >> 16) | ((unsigned)tile_w << 16));// td1[31:16], tile_dim0
  g1[4] = (int)(unsigned)tile_h;                        // tile_dim1 (tile_dim2=0)
  g1[5] = (int)(unsigned)stride_elems;                  // tensor_dim0_stride[31:0]
  g1[6] = 0; g1[7] = 0;
  i32x4 z4; z4[0]=0; z4[1]=0; z4[2]=0; z4[3]=0;
  i32x8 z8; for (int i = 0; i < 8; i++) z8[i] = 0;
  __builtin_amdgcn_tensor_load_to_lds(g0, g1, z4, z4, z8, 0);
}

// ---------------------------------------------------------------------------
// 1) Materialize Hamilton-product weight matrices (bf16) + convert others.
// ---------------------------------------------------------------------------
__global__ void k_build_weights(
    const float* qr, const float* qi, const float* qj, const float* qk,
    const float* kr, const float* ki, const float* kj, const float* kk,
    const float* vw, const float* ow, const float* evw,
    bf16_t* Wq, bf16_t* Wk, bf16_t* Wv, bf16_t* Wo, bf16_t* Wev)
{
  const int   compT[4][4] = {{0,1,2,3},{1,0,3,2},{2,3,0,1},{3,2,1,0}};
  const float sgnT[4][4]  = {{1.f,-1.f,-1.f,-1.f},{1.f,1.f,-1.f,1.f},
                             {1.f,1.f,1.f,-1.f},{1.f,-1.f,1.f,1.f}};
  const int total = 1572864;
  for (int e = blockIdx.x * blockDim.x + threadIdx.x; e < total;
       e += gridDim.x * blockDim.x) {
    if (e < 524288) {                       // Wq / Wk
      int which = e >> 18;
      int t = e & 262143;
      int o = t >> 9, d = t & 511;
      int bi = o >> 7, a = o & 127, bj = d >> 7, c = d & 127;
      int cmp = compT[bi][bj];
      const float* src;
      if (which == 0) src = (cmp==0)?qr:(cmp==1)?qi:(cmp==2)?qj:qk;
      else            src = (cmp==0)?kr:(cmp==1)?ki:(cmp==2)?kj:kk;
      float v = sgnT[bi][bj] * src[a*128 + c];
      ((which==0)?Wq:Wk)[t] = (bf16_t)v;
    } else if (e < 1048576) {               // Wv [512,1024]
      int t = e - 524288; Wv[t] = (bf16_t)vw[t];
    } else {
      int t = e - 1048576;
      if (t < 262144) Wo[t] = (bf16_t)ow[t];
      else            Wev[t-262144] = (bf16_t)evw[t-262144];
    }
  }
}

// ---------------------------------------------------------------------------
// 2) Activation prep + per-key threshold MLP.
// ---------------------------------------------------------------------------
__global__ void k_prep(
    const float* query, const float* key, const float* value, const float* cross,
    const float* dens, const float* thr, const float* dm1w, const float* dm1b,
    const float* dm2w, const float* dm2b,
    bf16_t* qin, bf16_t* ksum, bf16_t* vc, float* thrk)
{
  const int total = BN * 1024;
  for (int e = blockIdx.x * blockDim.x + threadIdx.x; e < total;
       e += gridDim.x * blockDim.x) {
    int row = e >> 10, col = e & 1023;
    float vv = (col < 512) ? value[row*512 + col] : cross[row*512 + (col-512)];
    vc[e] = (bf16_t)vv;
    if (col < 512) {
      int o = row*512 + col;
      qin[o]  = (bf16_t)query[o];
      ksum[o] = (bf16_t)(key[o] + cross[o]);
    }
    if (col == 0) {
      const float* feat = dens + (long)row*3;
      float dsh = dm2b[0];
      for (int o = 0; o < 16; o++) {
        float z = dm1b[o];
        for (int i2 = 0; i2 < 3; i2++) z += dm1w[o*3+i2] * feat[i2];
        float ge = 0.5f * z * (1.f + erff(z * 0.70710678118f));
        dsh += dm2w[o] * ge;
      }
      float tb = thr[0];
      float sp = (tb > 20.f) ? tb : log1pf(expf(tb));
      thrk[row] = sp + 0.1f * dsh;
    }
  }
}

// ---------------------------------------------------------------------------
// 3) bf16 WMMA GEMM: C[m,n] = sum_k A[m,k]*W[n,k] + bscale*bias[n]
//    A tile DMA'd by the TDM (padded rows land directly in As[64][40]).
//    W tile staged transposed [k][n]; B-fragments are read as strided 16-bit elements.
// ---------------------------------------------------------------------------
__global__ __launch_bounds__(128) void k_gemm(
    const bf16_t* __restrict__ A, const bf16_t* __restrict__ W,
    const float* __restrict__ bias, float bscale,
    float* Cf, bf16_t* Cb, int Kdim, int Nout, int outBF)
{
  __shared__ __align__(16) bf16_t As[64][40];  // 32 data + 8 pad per row
  __shared__ __align__(16) bf16_t Ws[32][72];  // transposed [k][n]
  const int tid  = threadIdx.x;
  const int wv   = tid >> 5, lane = tid & 31;
  const int g    = lane >> 4, r = lane & 15;
  const int mblk = blockIdx.x * 64;
  const int n0   = blockIdx.y * 64;
  const unsigned asOff = (unsigned)(size_t)&As[0][0];

  v8f acc[4];
  for (int q = 0; q < 4; q++) for (int i = 0; i < 8; i++) acc[q][i] = 0.f;

  for (int kk0 = 0; kk0 < Kdim; kk0 += 32) {
    __syncthreads();
    // A tile 64x32 via Tensor Data Mover: row = 16 DWORDs (code 3),
    // pad 4 DWORDs (code 3) -> 40-element LDS row stride.
    if (wv == 0)
      tdm_load_2d(A + (long)mblk*Kdim + kk0, asOff, 32, 64, Kdim, 3, 3);
    // W tile transposed: batch global loads into regs, then LDS stores
    bf16_t wtmp[16];
    for (int i = 0; i < 16; i++) {
      int u = tid + i*128;
      int n = u >> 5, k = u & 31;
      wtmp[i] = W[(long)(n0+n)*Kdim + kk0 + k];
    }
    for (int i = 0; i < 16; i++) {
      int u = tid + i*128;
      int n = u >> 5, k = u & 31;
      Ws[k][n] = wtmp[i];
    }
    if (kk0 + 32 < Kdim)   // gfx1250 global_prefetch_b8 of next W tile
      __builtin_prefetch(W + (long)(n0 + (tid>>1))*Kdim + kk0 + 32, 0, 1);
    if (wv == 0) __builtin_amdgcn_s_wait_tensorcnt(0);
    __syncthreads();

    // A fragment (16x32 bf16): two contiguous 8-elem runs at K = 8g, 16+8g
    v8bf lo = *(const v8bf*)&As[wv*16 + r][8*g];
    v8bf hi = *(const v8bf*)&As[wv*16 + r][16 + 8*g];
    v16bf af;
    for (int i = 0; i < 8; i++) { af[i] = lo[i]; af[i+8] = hi[i]; }

    const int kl = r + 16*g;
    for (int q = 0; q < 4; q++) {
      v16bf bq;
      for (int i = 0; i < 8; i++) { bq[i] = Ws[8*g + i][q*16 + r]; bq[i+8] = Ws[16 + 8*g + i][q*16 + r]; }
      acc[q] = __builtin_amdgcn_wmma_f32_16x16x32_bf16(
          false, af, false, bq, (short)0, acc[q], false, false);
    }
  }

  for (int q = 0; q < 4; q++) {
    int n = n0 + q*16 + r;
    float bv = bscale * bias[n];
    for (int v = 0; v < 8; v++) {
      int m = mblk + wv*16 + v + 8*g;
      float val = acc[q][v] + bv;
      long off = (long)m * Nout + n;
      if (outBF) Cb[off] = (bf16_t)val; else Cf[off] = val;
    }
  }
}

// ---------------------------------------------------------------------------
// 4) Flash attention with spike gating. WG = 64 query rows of one (b,h);
//    key blocks of 32. V tile via TDM; K tile staged transposed manually.
// ---------------------------------------------------------------------------
__global__ __launch_bounds__(128) void k_flash(
    const bf16_t* __restrict__ Q, const bf16_t* __restrict__ K,
    const bf16_t* __restrict__ V, const float* __restrict__ thrk,
    const float* __restrict__ mask, bf16_t* __restrict__ attnb)
{
  __shared__ __align__(16) bf16_t Ks[64][40];      // [d][n] transposed K tile
  __shared__ __align__(16) bf16_t Vs[32][72];      // [n][d] V tile (TDM, padded)
  __shared__ __align__(16) bf16_t Ps[4][16][40];   // per-wave P transpose scratch
  __shared__ float Sth[32];
  __shared__ float Skm[32];

  const int tid = threadIdx.x;
  const int wv = tid >> 5, lane = tid & 31, g = lane >> 4, r = lane & 15;
  const int b = blockIdx.z, h = blockIdx.y, mblk = blockIdx.x * 64;
  const unsigned vsOff = (unsigned)(size_t)&Vs[0][0];

  const long qbase = ((long)(b*NN + mblk + wv*16 + r)) * DD + h*HD;
  v16bf aq[2];
  for (int c = 0; c < 2; c++) {
    v8bf lo = *(const v8bf*)(Q + qbase + 32*c + 8*g);
    v8bf hi = *(const v8bf*)(Q + qbase + 32*c + 16 + 8*g);
    for (int i = 0; i < 8; i++) { aq[c][i] = lo[i]; aq[c][i+8] = hi[i]; }
  }
  float qm[8];
  for (int v = 0; v < 8; v++) qm[v] = mask[b*NN + mblk + wv*16 + v + 8*g];

  v8f O[4];
  for (int c = 0; c < 4; c++) for (int i = 0; i < 8; i++) O[c][i] = 0.f;
  float runm[8], runl[8];
  for (int v = 0; v < 8; v++) { runm[v] = -3.0e38f; runl[v] = 0.f; }

  const int kl = r + 16*g;

  for (int j = 0; j < NN; j += 32) {
    __syncthreads();
    // V tile 32x64 via TDM: row = 32 DWORDs (code 4), pad 4 DWORDs (code 3)
    // -> 72-element LDS row stride.
    if (wv == 0)
      tdm_load_2d(V + ((long)(b*NN + j))*DD + h*HD, vsOff, 64, 32, DD, 4, 3);
    // K tile transposed [d][n]: regs first, then LDS
    bf16_t ktmp[16];
    for (int i = 0; i < 16; i++) {
      int u = tid + i*128;
      int n = u >> 6, d = u & 63;
      ktmp[i] = K[((long)(b*NN + j + n)) * DD + h*HD + d];
    }
    for (int i = 0; i < 16; i++) {
      int u = tid + i*128;
      int n = u >> 6, d = u & 63;
      Ks[d][n] = ktmp[i];
    }
    if (tid < 32) { Sth[tid] = thrk[b*NN + j + tid]; Skm[tid] = mask[b*NN + j + tid]; }
    if (wv == 0) __builtin_amdgcn_s_wait_tensorcnt(0);
    __syncthreads();

    // S = Q x K^T
    v8f s[2];
    for (int q = 0; q < 2; q++) for (int i = 0; i < 8; i++) s[q][i] = 0.f;
    for (int q = 0; q < 2; q++)
      for (int c = 0; c < 2; c++) {
        v16bf bq;
        for (int i = 0; i < 8; i++) { bq[i] = Ks[32*c + 8*g + i][q*16 + r]; bq[i+8] = Ks[32*c + 16 + 8*g + i][q*16 + r]; }
        s[q] = __builtin_amdgcn_wmma_f32_16x16x32_bf16(
            false, aq[c], false, bq, (short)0, s[q], false, false);
      }

    // scale, clip, spike gate, mask, clip; online softmax
    float pv[2][8], mloc[8];
    for (int v = 0; v < 8; v++) {
      float mrow = -3.4e38f;
      for (int q = 0; q < 2; q++) {
        float sv = s[q][v] * 0.25f;                      // 1/sqrt(QC=16)
        sv = fminf(fmaxf(sv, -6.f), 6.f);
        float th   = Sth[q*16 + r];
        float gate = 1.f / (1.f + __expf(-5.f * (sv - th)));
        float mod  = sv * (1.f + 2.f * gate);
        if (qm[v] == 0.f || Skm[q*16 + r] == 0.f) mod = -10000.f;
        mod = fminf(fmaxf(mod, -30.f), 30.f);
        pv[q][v] = mod;
        mrow = fmaxf(mrow, mod);
      }
      mloc[v] = wred_max16(mrow);
    }
    for (int v = 0; v < 8; v++) {
      float nm = fmaxf(runm[v], mloc[v]);
      float alpha = __expf(runm[v] - nm);
      runm[v] = nm;
      float ps = 0.f;
      for (int q = 0; q < 2; q++) {
        float p = __expf(pv[q][v] - nm);
        pv[q][v] = p; ps += p;
      }
      ps = wred_sum16(ps);
      runl[v] = runl[v] * alpha + ps;
      for (int c = 0; c < 4; c++) O[c][v] *= alpha;
    }

    // C-layout -> A-layout transpose of P through wave-private LDS
    for (int v = 0; v < 8; v++)
      for (int q = 0; q < 2; q++)
        Ps[wv][v + 8*g][q*16 + r] = (bf16_t)pv[q][v];
    v16bf ap;
    {
      v8bf lo = *(const v8bf*)&Ps[wv][r][8*g];
      v8bf hi = *(const v8bf*)&Ps[wv][r][16 + 8*g];
      for (int i = 0; i < 8; i++) { ap[i] = lo[i]; ap[i+8] = hi[i]; }
    }
    // O += P x V
    for (int c = 0; c < 4; c++) {
      v16bf bv;
      for (int i = 0; i < 8; i++) { bv[i] = Vs[8*g + i][c*16 + r]; bv[i+8] = Vs[16 + 8*g + i][c*16 + r]; }
      O[c] = __builtin_amdgcn_wmma_f32_16x16x32_bf16(
          false, ap, false, bv, (short)0, O[c], false, false);
    }
  }

  for (int c = 0; c < 4; c++)
    for (int v = 0; v < 8; v++) {
      float val = O[c][v] / runl[v];
      long off = ((long)(b*NN + mblk + wv*16 + v + 8*g)) * DD + h*HD + c*16 + r;
      attnb[off] = (bf16_t)val;
    }
}

// ---------------------------------------------------------------------------
// 5) Stable argsort via bitonic sort of composite key (time*2048 + idx).
// ---------------------------------------------------------------------------
__global__ __launch_bounds__(1024) void k_sort(const int* time_idx, int* order)
{
  __shared__ unsigned keys[2048];
  const int b = blockIdx.x, t = threadIdx.x;
  for (int rep = 0; rep < 2; rep++) {
    int i = t + rep*1024;
    keys[i] = ((unsigned)time_idx[b*NN + i]) * 2048u + (unsigned)i;
  }
  __syncthreads();
  for (unsigned k = 2; k <= 2048; k <<= 1)
    for (unsigned jj = k >> 1; jj > 0; jj >>= 1) {
      for (int rep = 0; rep < 2; rep++) {
        unsigned i = (unsigned)(t + rep*1024);
        unsigned ixj = i ^ jj;
        if (ixj > i) {
          unsigned a = keys[i], c = keys[ixj];
          bool up = ((i & k) == 0);
          if ((a > c) == up) { keys[i] = c; keys[ixj] = a; }
        }
      }
      __syncthreads();
    }
  for (int rep = 0; rep < 2; rep++) {
    int i = t + rep*1024;
    order[b*NN + i] = (int)(keys[i] & 2047u);
  }
}

// ---------------------------------------------------------------------------
// 6) Temporal event conv: windowed filter match + softmax over K=5 -> ctx.
// ---------------------------------------------------------------------------
__global__ __launch_bounds__(128) void k_event(
    const float* __restrict__ query, const float* __restrict__ mask,
    const int* __restrict__ order, const float* __restrict__ ef,
    const float* __restrict__ ec, bf16_t* __restrict__ ctxb)
{
  const int gw = blockIdx.x * 4 + (threadIdx.x >> 5);
  const int lane = threadIdx.x & 31;
  const int b = gw >> 11, j = gw & 2047;
  const int d0 = lane * 16;

  float acc[5] = {0.f, 0.f, 0.f, 0.f, 0.f};
  for (int w = 0; w < 5; w++) {
    int pos = j + w - 2;
    if (pos < 0 || pos >= NN) continue;
    int orig = order[b*NN + pos];
    float qv[16];
    for (int i = 0; i < 16; i++) qv[i] = query[((long)(b*NN+orig))*DD + d0 + i];
    for (int k = 0; k < 5; k++) {
      float p = 0.f;
      for (int i = 0; i < 16; i++) p += qv[i] * ef[(k*5 + w)*DD + d0 + i];
      acc[k] += p;
    }
  }
  for (int k = 0; k < 5; k++)
    acc[k] = wred_sum32(acc[k]) * 0.08838834764831845f; // 1/sqrt(QI=128)

  float ms = mask[b*NN + order[b*NN + j]];
  if (ms == 0.f) for (int k = 0; k < 5; k++) acc[k] = -10000.f;
  float mx = acc[0];
  for (int k = 1; k < 5; k++) mx = fmaxf(mx, acc[k]);
  float den = 0.f;
  for (int k = 0; k < 5; k++) { acc[k] = __expf(acc[k] - mx); den += acc[k]; }
  for (int k = 0; k < 5; k++) acc[k] /= den;

  for (int i = 0; i < 16; i++) {
    float sv = 0.f;
    for (int k = 0; k < 5; k++) sv += acc[k] * ec[k*DD + d0 + i];
    ctxb[((long)(b*NN + j))*DD + d0 + i] = (bf16_t)sv;
  }
}

// ---------------------------------------------------------------------------
// 7) Finalize: layernorm(ev_out) * ms, inverse-permute scatter-add into out.
// ---------------------------------------------------------------------------
__global__ __launch_bounds__(128) void k_final(
    const float* __restrict__ evout, const float* __restrict__ mask,
    const int* __restrict__ order, const float* __restrict__ lng,
    const float* __restrict__ lnb, float* __restrict__ out)
{
  const int gw = blockIdx.x * 4 + (threadIdx.x >> 5);
  const int lane = threadIdx.x & 31;
  const int b = gw >> 11, j = gw & 2047;
  const int d0 = lane * 16;

  float x[16];
  float s = 0.f, s2 = 0.f;
  for (int i = 0; i < 16; i++) {
    x[i] = evout[((long)(b*NN + j))*DD + d0 + i];
    s += x[i]; s2 += x[i]*x[i];
  }
  s = wred_sum32(s); s2 = wred_sum32(s2);
  float mean = s * (1.f/512.f);
  float var  = s2 * (1.f/512.f) - mean*mean;
  float rinv = rsqrtf(var + 1e-5f);
  int orig = order[b*NN + j];
  float ms = mask[b*NN + orig];
  float scale = ms * ms;
  for (int i = 0; i < 16; i++) {
    int d = d0 + i;
    float y = (x[i] - mean) * rinv * lng[d] + lnb[d];
    out[((long)(b*NN + orig))*DD + d] += y * scale;
  }
}

// ---------------------------------------------------------------------------
extern "C" void kernel_launch(void* const* d_in, const int* in_sizes, int n_in,
                              void* d_out, int out_size, void* d_ws, size_t ws_size,
                              hipStream_t stream)
{
  const float* query = (const float*)d_in[0];
  const float* key   = (const float*)d_in[1];
  const float* value = (const float*)d_in[2];
  const float* cross = (const float*)d_in[3];
  const float* dens  = (const float*)d_in[4];
  const float* mask  = (const float*)d_in[5];
  const int*   timei = (const int*)  d_in[6];
  const float* q_r = (const float*)d_in[7],  *q_i = (const float*)d_in[8];
  const float* q_j = (const float*)d_in[9],  *q_k = (const float*)d_in[10];
  const float* q_b = (const float*)d_in[11];
  const float* k_r = (const float*)d_in[12], *k_i = (const float*)d_in[13];
  const float* k_j = (const float*)d_in[14], *k_k = (const float*)d_in[15];
  const float* k_b = (const float*)d_in[16];
  const float* v_w = (const float*)d_in[17], *v_b = (const float*)d_in[18];
  const float* o_w = (const float*)d_in[19], *o_b = (const float*)d_in[20];
  const float* thr = (const float*)d_in[21];
  const float* dm1w = (const float*)d_in[22], *dm1b = (const float*)d_in[23];
  const float* dm2w = (const float*)d_in[24], *dm2b = (const float*)d_in[25];
  const float* ef  = (const float*)d_in[26], *ec  = (const float*)d_in[27];
  const float* evw = (const float*)d_in[28], *evb = (const float*)d_in[29];
  const float* lng = (const float*)d_in[30], *lnb = (const float*)d_in[31];
  float* out = (float*)d_out;

  char* ws = (char*)d_ws;
  size_t off = 0;
  auto alloc = [&](size_t bytes) -> void* {
    void* p = ws + off; off += (bytes + 255) & ~(size_t)255; return p;
  };
  bf16_t* Wq   = (bf16_t*)alloc(512*512*2);
  bf16_t* Wk   = (bf16_t*)alloc(512*512*2);
  bf16_t* Wv   = (bf16_t*)alloc((size_t)512*1024*2);
  bf16_t* Wo   = (bf16_t*)alloc(512*512*2);
  bf16_t* Wev  = (bf16_t*)alloc(512*512*2);
  bf16_t* qin  = (bf16_t*)alloc((size_t)BN*512*2);
  bf16_t* ksum = (bf16_t*)alloc((size_t)BN*512*2);
  bf16_t* vc   = (bf16_t*)alloc((size_t)BN*1024*2);
  bf16_t* qp   = (bf16_t*)alloc((size_t)BN*512*2);
  bf16_t* kp   = (bf16_t*)alloc((size_t)BN*512*2);
  bf16_t* vp   = (bf16_t*)alloc((size_t)BN*512*2);
  bf16_t* attnb= (bf16_t*)alloc((size_t)BN*512*2);
  bf16_t* ctxb = (bf16_t*)alloc((size_t)BN*512*2);
  float*  thrk = (float*) alloc((size_t)BN*4);
  int*    order= (int*)   alloc((size_t)BN*4);
  float*  evout= (float*) alloc((size_t)BN*512*4);

  k_build_weights<<<1024, 256, 0, stream>>>(q_r,q_i,q_j,q_k, k_r,k_i,k_j,k_k,
                                            v_w, o_w, evw, Wq, Wk, Wv, Wo, Wev);
  k_prep<<<2048, 256, 0, stream>>>(query, key, value, cross, dens, thr,
                                   dm1w, dm1b, dm2w, dm2b, qin, ksum, vc, thrk);

  dim3 gg(BN/64, 512/64);
  k_gemm<<<gg, 128, 0, stream>>>(qin,  Wq, q_b, 1.0f, nullptr, qp, 512, 512, 1);
  // qlinear(key)+qlinear(cross) == qlinear(key+cross), bias x2
  k_gemm<<<gg, 128, 0, stream>>>(ksum, Wk, k_b, 2.0f, nullptr, kp, 512, 512, 1);
  k_gemm<<<gg, 128, 0, stream>>>(vc,   Wv, v_b, 1.0f, nullptr, vp, 1024, 512, 1);

  k_flash<<<dim3(NN/64, HH, BB), 128, 0, stream>>>(qp, kp, vp, thrk, mask, attnb);

  k_sort<<<BB, 1024, 0, stream>>>(timei, order);
  k_event<<<BN/4, 128, 0, stream>>>(query, mask, order, ef, ec, ctxb);
  k_gemm<<<gg, 128, 0, stream>>>(ctxb, Wev, evb, 1.0f, evout, nullptr, 512, 512, 0);
  k_gemm<<<gg, 128, 0, stream>>>(attnb, Wo, o_b, 1.0f, out, nullptr, 512, 512, 0);
  k_final<<<BN/4, 128, 0, stream>>>(evout, mask, order, lng, lnb, out);
}
